// AttBiLstmModel_40295383171633
// MI455X (gfx1250) — matrix operand order repaired and checked by reading, not compiled, not run on hardware
//
#include <hip/hip_runtime.h>
#include <hip/hip_bf16.h>

#define BB 64
#define TT 2048
#define DD 128
#define HH 256
#define FH 1024          // 4*H
#define KT_TILES 12      // 384/32 total K tiles ([rec;k] concatenated)
#define NT_TILES 64      // 1024/16
#define AROW 512         // LDS A row: 256 h | 128 x buf0 | 128 x buf1

typedef __attribute__((ext_vector_type(16))) __bf16 v16bf;
typedef __attribute__((ext_vector_type(8)))  float  v8f;

union Frag16 { v16bf v; uint4 q[2]; };
union F4u    { float4 v; float f[4]; };

__device__ __forceinline__ unsigned short f2bf(float f) {
    unsigned int u = __float_as_uint(f);
    u += 0x7FFFu + ((u >> 16) & 1u);          // round-to-nearest-even
    return (unsigned short)(u >> 16);
}
__device__ __forceinline__ float bf2f(unsigned short s) {
    return __uint_as_float(((unsigned int)s) << 16);
}
__device__ __forceinline__ float bfu_lo(unsigned int u) { return __uint_as_float(u << 16); }
__device__ __forceinline__ float bfu_hi(unsigned int u) { return __uint_as_float(u & 0xFFFF0000u); }

__device__ __forceinline__ float fsig(float x) {
    float e = __builtin_amdgcn_exp2f(-1.44269504f * x);           // v_exp_f32
    return __builtin_amdgcn_rcpf(1.0f + e);                       // v_rcp_f32
}
__device__ __forceinline__ float ftanh(float x) {
    x = fminf(20.0f, fmaxf(-20.0f, x));
    float e = __builtin_amdgcn_exp2f(-2.88539008f * x);
    return (1.0f - e) * __builtin_amdgcn_rcpf(1.0f + e);
}

// ---------------------------------------------------------------------------
// Kernel 1: build bf16 [rec;k] (K=384 x N=1024) per direction, pre-swizzled
// into WMMA B-fragment order: block(kt,nt) -> 32 lanes x 16 bf16 contiguous.
// ---------------------------------------------------------------------------
__global__ void prep_weights(const float* __restrict__ k_fwd,
                             const float* __restrict__ r_fwd,
                             const float* __restrict__ k_bwd,
                             const float* __restrict__ r_bwd,
                             unsigned short* __restrict__ wpF,
                             unsigned short* __restrict__ wpB) {
    const int per_dir = KT_TILES * NT_TILES * 32 * 16;            // 393216
    int g = blockIdx.x * blockDim.x + threadIdx.x;
    if (g >= 2 * per_dir) return;
    int dir = g / per_dir;
    int idx = g - dir * per_dir;
    int kk    = idx & 15;
    int lane  = (idx >> 4) & 31;
    int block = idx >> 9;                                          // kt*64 + nt
    int kt = block >> 6, nt = block & 63;
    int k = kt * 32 + (lane >> 4) * 16 + kk;
    int n = nt * 16 + (lane & 15);
    const float* rec = dir ? r_bwd : r_fwd;
    const float* km  = dir ? k_bwd : k_fwd;
    float v = (k < HH) ? rec[k * FH + n] : km[(k - HH) * FH + n];
    unsigned short* dst = dir ? wpB : wpF;
    dst[idx] = f2bf(v);
}

// ---------------------------------------------------------------------------
// Kernel 1b: x fp32 -> bf16 (so recurrence can async-copy rows straight to LDS)
// ---------------------------------------------------------------------------
__global__ void prep_x(const float* __restrict__ x, unsigned short* __restrict__ xbf) {
    int g = blockIdx.x * blockDim.x + threadIdx.x;                 // 1 thread = 8 elems
    const float4* xp = reinterpret_cast<const float4*>(x) + (size_t)g * 2;
    float4 a0 = xp[0], a1 = xp[1];
    uint4 u;
    u.x = (unsigned)f2bf(a0.x) | ((unsigned)f2bf(a0.y) << 16);
    u.y = (unsigned)f2bf(a0.z) | ((unsigned)f2bf(a0.w) << 16);
    u.z = (unsigned)f2bf(a1.x) | ((unsigned)f2bf(a1.y) << 16);
    u.w = (unsigned)f2bf(a1.z) | ((unsigned)f2bf(a1.w) << 16);
    reinterpret_cast<uint4*>(xbf)[g] = u;
}

// ---------------------------------------------------------------------------
// Kernel 2: persistent BiLSTM recurrence. grid = 8 WGs (dir*4 + btile),
// 256 threads = 8 waves. Each WG owns 16 batch rows for all T steps.
// Weights stream from L2 (opaque byte offset keeps them as in-loop
// global_load_b128); x_t is double-buffered into LDS via async-to-LDS.
// ---------------------------------------------------------------------------
__global__ void __launch_bounds__(256)
bilstm_kernel(const unsigned short* __restrict__ xbf,
              const unsigned short* __restrict__ wpF,
              const unsigned short* __restrict__ wpB,
              const float* __restrict__ b_fwd,
              const float* __restrict__ b_bwd,
              unsigned short* __restrict__ hF,
              unsigned short* __restrict__ hB) {
    __shared__ __align__(16) unsigned short sA[16 * AROW];        // [m][ h(256) | x0(128) | x1(128) ]
    __shared__ __align__(16) float sZ[16 * FH];                   // [m][n] f32 gates

    const int dir = blockIdx.x >> 2;
    const int b0  = (blockIdx.x & 3) * 16;
    const unsigned short* W = dir ? wpB : wpF;
    const float* bias       = dir ? b_bwd : b_fwd;
    unsigned short* hout    = dir ? hB : hF;

    const int tid  = threadIdx.x;
    const int wave = tid >> 5, lane = tid & 31;
    const int lhi  = lane >> 4, llo = lane & 15;

    float biasr[8];
#pragma unroll
    for (int it = 0; it < 8; ++it)
        biasr[it] = bias[(wave * 8 + it) * 16 + llo];

    // zero h region of A (cols 0..255), zero cell state
    for (int e = tid; e < 16 * HH; e += 256) {
        int m = e >> 8, j = e & 255;
        sA[m * AROW + j] = 0;
    }
    float c_reg[16];
#pragma unroll
    for (int k = 0; k < 16; ++k) c_reg[k] = 0.0f;

    const int m_s = tid >> 4, d0 = (tid & 15) * 8;                // x staging ids (16B/thread)
    const int m_e = tid >> 4, jb = (tid & 15) * 16;               // elementwise ids

    // prefetch x for t=0 into buf0 (ASYNCcnt = 1 outstanding)
    {
        const int tt0 = dir ? (TT - 1) : 0;
        const unsigned short* gp = xbf + ((size_t)(b0 + m_s) * TT + tt0) * DD + d0;
        unsigned lds_off = (unsigned)(size_t)(&sA[m_s * AROW + HH + d0]);
        asm volatile("global_load_async_to_lds_b128 %0, %1, off"
                     :: "v"(lds_off), "v"(gp) : "memory");
    }

    unsigned int woff = 0;                                        // opaque 0: defeats hoisting,
    for (int t = 0; t < TT; ++t) {                                // keeps W's global provenance
        asm volatile("" : "+s"(woff));
        const int tt = dir ? (TT - 1 - t) : t;

        // ---- issue async x prefetch for step t+1 into alternate buffer ----
        {
            const int ts = (t + 1 < TT) ? (t + 1) : (TT - 1);
            const int tts = dir ? (TT - 1 - ts) : ts;
            const int xsel_n = (t + 1) & 1;
            const unsigned short* gp = xbf + ((size_t)(b0 + m_s) * TT + tts) * DD + d0;
            unsigned lds_off = (unsigned)(size_t)(&sA[m_s * AROW + HH + xsel_n * DD + d0]);
            asm volatile("global_load_async_to_lds_b128 %0, %1, off"
                         :: "v"(lds_off), "v"(gp) : "memory");
        }
        // oldest async (this step's x) complete; 1 (the prefetch) may remain
        asm volatile("s_wait_asynccnt 0x1" ::: "memory");
        __syncthreads();

        // ---- z = [h|x_t] @ [rec;k] + bias  via WMMA bf16 ----
        v8f acc[8];
#pragma unroll
        for (int it = 0; it < 8; ++it) {
            v8f a;
#pragma unroll
            for (int r = 0; r < 8; ++r) a[r] = biasr[it];
            acc[it] = a;
        }
        const int xsel = t & 1;
        // per-(kt,nt) block = 32 lanes * 16 bf16 = 64 uint4; lane offset = 2 uint4
        const uint4* wq = reinterpret_cast<const uint4*>(
                              reinterpret_cast<const char*>(W) + woff) + lane * 2;
#pragma unroll 2
        for (int kt = 0; kt < KT_TILES; ++kt) {
            const int col = (kt < 8) ? (kt * 32) : (HH + xsel * DD + (kt - 8) * 32);
            Frag16 A;
            const int a0i = llo * AROW + col + lhi * 16;
            A.q[0] = *reinterpret_cast<const uint4*>(&sA[a0i]);
            A.q[1] = *reinterpret_cast<const uint4*>(&sA[a0i + 8]);

            const uint4* bp = wq + (size_t)(kt * NT_TILES + wave * 8) * 64;
            Frag16 Bf[8];
#pragma unroll
            for (int it = 0; it < 8; ++it) {                      // one clause of 16 loads
                Bf[it].q[0] = bp[it * 64 + 0];
                Bf[it].q[1] = bp[it * 64 + 1];
            }
#pragma unroll
            for (int it = 0; it < 8; ++it) {
                acc[it] = __builtin_amdgcn_wmma_f32_16x16x32_bf16(
                    false, A.v, false, Bf[it].v, (short)0, acc[it], false, false);
            }
        }
        // write z tiles to LDS [m][n]
#pragma unroll
        for (int it = 0; it < 8; ++it) {
            const int n = (wave * 8 + it) * 16 + llo;
            const int mb = lhi * 8;
#pragma unroll
            for (int r = 0; r < 8; ++r)
                sZ[(mb + r) * FH + n] = acc[it][r];
        }
        __syncthreads();

        // ---- gates: c,h update (thread owns row m_e, cols jb..jb+15) ----
        {
            const float* zr = sZ + m_e * FH;
            const float4* ZI = reinterpret_cast<const float4*>(zr + jb);
            const float4* ZF = reinterpret_cast<const float4*>(zr + 256 + jb);
            const float4* ZG = reinterpret_cast<const float4*>(zr + 512 + jb);
            const float4* ZO = reinterpret_cast<const float4*>(zr + 768 + jb);
            float hv[16];
#pragma unroll
            for (int q = 0; q < 4; ++q) {
                F4u zi, zf, zg, zo;
                zi.v = ZI[q]; zf.v = ZF[q]; zg.v = ZG[q]; zo.v = ZO[q];
#pragma unroll
                for (int e = 0; e < 4; ++e) {
                    const int k = q * 4 + e;
                    float c = fsig(zf.f[e]) * c_reg[k] + fsig(zi.f[e]) * ftanh(zg.f[e]);
                    c_reg[k] = c;
                    hv[k] = fsig(zo.f[e]) * ftanh(c);
                }
            }
            uint4 u0, u1;
            u0.x = (unsigned)f2bf(hv[0])  | ((unsigned)f2bf(hv[1])  << 16);
            u0.y = (unsigned)f2bf(hv[2])  | ((unsigned)f2bf(hv[3])  << 16);
            u0.z = (unsigned)f2bf(hv[4])  | ((unsigned)f2bf(hv[5])  << 16);
            u0.w = (unsigned)f2bf(hv[6])  | ((unsigned)f2bf(hv[7])  << 16);
            u1.x = (unsigned)f2bf(hv[8])  | ((unsigned)f2bf(hv[9])  << 16);
            u1.y = (unsigned)f2bf(hv[10]) | ((unsigned)f2bf(hv[11]) << 16);
            u1.z = (unsigned)f2bf(hv[12]) | ((unsigned)f2bf(hv[13]) << 16);
            u1.w = (unsigned)f2bf(hv[14]) | ((unsigned)f2bf(hv[15]) << 16);
            uint4* ap = reinterpret_cast<uint4*>(&sA[m_e * AROW + jb]);   // next-step A
            ap[0] = u0; ap[1] = u1;
            uint4* hp = reinterpret_cast<uint4*>(
                hout + ((size_t)(b0 + m_e) * TT + tt) * HH + jb);          // h -> global
            hp[0] = u0; hp[1] = u1;
        }
        __syncthreads();
    }
}

// ---------------------------------------------------------------------------
// Kernel 3: attention pooling. grid = B, block = 256 (== H).
// ---------------------------------------------------------------------------
__global__ void __launch_bounds__(256)
attn_kernel(const unsigned short* __restrict__ hF,
            const unsigned short* __restrict__ hB,
            const float* __restrict__ att_w,
            float* __restrict__ out) {
    __shared__ float sW[HH];
    __shared__ float sL[TT];
    __shared__ float sR[256];
    const int b = blockIdx.x, tid = threadIdx.x;
    sW[tid] = att_w[tid];
    __syncthreads();

    for (int t = tid; t < TT; t += 256) {
        const uint4* pf = reinterpret_cast<const uint4*>(hF + ((size_t)b * TT + t) * HH);
        const uint4* pb = reinterpret_cast<const uint4*>(hB + ((size_t)b * TT + t) * HH);
        float dot = 0.0f;
#pragma unroll 4
        for (int h8 = 0; h8 < HH / 8; ++h8) {
            uint4 qa = pf[h8], qb = pb[h8];
            const int h0 = h8 * 8;
            dot += sW[h0 + 0] * ftanh(bfu_lo(qa.x) + bfu_lo(qb.x));
            dot += sW[h0 + 1] * ftanh(bfu_hi(qa.x) + bfu_hi(qb.x));
            dot += sW[h0 + 2] * ftanh(bfu_lo(qa.y) + bfu_lo(qb.y));
            dot += sW[h0 + 3] * ftanh(bfu_hi(qa.y) + bfu_hi(qb.y));
            dot += sW[h0 + 4] * ftanh(bfu_lo(qa.z) + bfu_lo(qb.z));
            dot += sW[h0 + 5] * ftanh(bfu_hi(qa.z) + bfu_hi(qb.z));
            dot += sW[h0 + 6] * ftanh(bfu_lo(qa.w) + bfu_lo(qb.w));
            dot += sW[h0 + 7] * ftanh(bfu_hi(qa.w) + bfu_hi(qb.w));
        }
        sL[t] = dot;
    }
    __syncthreads();

    float mx = -3.4e38f;
    for (int t = tid; t < TT; t += 256) mx = fmaxf(mx, sL[t]);
    sR[tid] = mx; __syncthreads();
    for (int s = 128; s > 0; s >>= 1) {
        if (tid < s) sR[tid] = fmaxf(sR[tid], sR[tid + s]);
        __syncthreads();
    }
    const float gmax = sR[0];
    __syncthreads();
    float lsum = 0.0f;
    for (int t = tid; t < TT; t += 256) {
        float e = __builtin_amdgcn_exp2f((sL[t] - gmax) * 1.44269504f);
        sL[t] = e; lsum += e;
    }
    sR[tid] = lsum; __syncthreads();
    for (int s = 128; s > 0; s >>= 1) {
        if (tid < s) sR[tid] += sR[tid + s];
        __syncthreads();
    }
    const float inv = __builtin_amdgcn_rcpf(sR[0]);
    __syncthreads();

    float r = 0.0f;
    const unsigned short* pf = hF + (size_t)b * TT * HH + tid;
    const unsigned short* pb = hB + (size_t)b * TT * HH + tid;
    for (int t = 0; t < TT; ++t)
        r += (sL[t] * inv) * (bf2f(pf[(size_t)t * HH]) + bf2f(pb[(size_t)t * HH]));
    out[b * HH + tid] = ftanh(r);
}

// ---------------------------------------------------------------------------
extern "C" void kernel_launch(void* const* d_in, const int* in_sizes, int n_in,
                              void* d_out, int out_size, void* d_ws, size_t ws_size,
                              hipStream_t stream) {
    (void)in_sizes; (void)n_in; (void)out_size; (void)ws_size;
    const float* x     = (const float*)d_in[0];
    const float* k_fwd = (const float*)d_in[1];
    const float* r_fwd = (const float*)d_in[2];
    const float* b_fwd = (const float*)d_in[3];
    const float* k_bwd = (const float*)d_in[4];
    const float* r_bwd = (const float*)d_in[5];
    const float* b_bwd = (const float*)d_in[6];
    const float* att_w = (const float*)d_in[7];
    float* out = (float*)d_out;

    char* ws = (char*)d_ws;
    const size_t wbytes = (size_t)(KT_TILES * 32) * FH * 2;       // 786432 B per dir
    unsigned short* wpF = (unsigned short*)(ws);
    unsigned short* wpB = (unsigned short*)(ws + wbytes);
    const size_t xbytes = (size_t)BB * TT * DD * 2;               // 32 MB bf16 x
    unsigned short* xbf = (unsigned short*)(ws + 2 * wbytes);
    const size_t hbytes = (size_t)BB * TT * HH * 2;               // 64 MB per dir
    unsigned short* hF = (unsigned short*)(ws + 2 * wbytes + xbytes);
    unsigned short* hB = (unsigned short*)(ws + 2 * wbytes + xbytes + hbytes);

    const int prep_elems = 2 * KT_TILES * NT_TILES * 32 * 16;     // 786432
    prep_weights<<<(prep_elems + 255) / 256, 256, 0, stream>>>(
        k_fwd, r_fwd, k_bwd, r_bwd, wpF, wpB);

    const int x8 = (BB * TT * DD) / 8;                            // 2,097,152 threads
    prep_x<<<x8 / 256, 256, 0, stream>>>(x, xbf);

    bilstm_kernel<<<8, 256, 0, stream>>>(xbf, wpF, wpB, b_fwd, b_bwd, hF, hB);

    attn_kernel<<<BB, 256, 0, stream>>>(hF, hB, att_w, out);
}
